// Layer_46514495815876
// MI455X (gfx1250) — matrix operand order repaired and checked by reading, not compiled, not run on hardware
//
#include <hip/hip_runtime.h>

// ---------------------------------------------------------------------------
// Types for CDNA5 WMMA (wave32)
// ---------------------------------------------------------------------------
typedef __attribute__((ext_vector_type(16))) _Float16 v16h;
typedef __attribute__((ext_vector_type(8)))  _Float16 v8h;
typedef __attribute__((ext_vector_type(4)))  _Float16 v4h;
typedef __attribute__((ext_vector_type(8)))  float    v8f;

union v16h_u { v16h v; v8h h[2]; };

// Problem dims (fixed by the reference)
static constexpr int M_DIM = 8192;
static constexpr int N_DIM = 4096;   // OUT
static constexpr int K_DIM = 4096;   // IN

// ---------------------------------------------------------------------------
// Kernel 1: per-row L2 norm of x, write x_norm as f16.
// ---------------------------------------------------------------------------
__global__ __launch_bounds__(256) void rownorm_f16_kernel(
    const float* __restrict__ x, _Float16* __restrict__ xh) {
  const size_t row = blockIdx.x;
  const float* xr = x + row * (size_t)K_DIM;
  _Float16* xo = xh + row * (size_t)K_DIM;

  float s = 0.0f;
#pragma unroll
  for (int i = 0; i < 4; ++i) {
    const int k = (threadIdx.x + i * 256) * 4;
    const float4 v = *(const float4*)(xr + k);
    s = fmaf(v.x, v.x, fmaf(v.y, v.y, fmaf(v.z, v.z, fmaf(v.w, v.w, s))));
  }
#pragma unroll
  for (int off = 16; off > 0; off >>= 1) s += __shfl_xor(s, off, 32);

  __shared__ float red[8];
  __shared__ float invs;
  const int wave = threadIdx.x >> 5;
  const int lane = threadIdx.x & 31;
  if (lane == 0) red[wave] = s;
  __syncthreads();
  if (threadIdx.x == 0) {
    float t = 0.0f;
#pragma unroll
    for (int i = 0; i < 8; ++i) t += red[i];
    invs = 1.0f / (sqrtf(t) + 1e-4f);   // x / (||x|| + eps)
  }
  __syncthreads();
  const float inv = invs;

#pragma unroll
  for (int i = 0; i < 4; ++i) {
    const int k = (threadIdx.x + i * 256) * 4;
    const float4 v = *(const float4*)(xr + k);
    v4h h;
    h[0] = (_Float16)(v.x * inv);
    h[1] = (_Float16)(v.y * inv);
    h[2] = (_Float16)(v.z * inv);
    h[3] = (_Float16)(v.w * inv);
    *(v4h*)(xo + k) = h;
  }
}

// ---------------------------------------------------------------------------
// Kernel 2: convert W (f32) -> f16.
// ---------------------------------------------------------------------------
__global__ __launch_bounds__(256) void wconv_f16_kernel(
    const float* __restrict__ W, _Float16* __restrict__ Wh) {
  const size_t k = ((size_t)blockIdx.x * 256 + threadIdx.x) * 4;
  const float4 v = *(const float4*)(W + k);
  v4h h;
  h[0] = (_Float16)v.x; h[1] = (_Float16)v.y;
  h[2] = (_Float16)v.z; h[3] = (_Float16)v.w;
  *(v4h*)(Wh + k) = h;
}

// ---------------------------------------------------------------------------
// Kernel 3: spike = (Xh @ Wh^T >= 1.0) via V_WMMA_F32_16X16X32_F16.
// Block = 256 threads = 8 waves (4 along M x 2 along N); wave tile 32x32,
// block tile 128(M) x 64(N). B tile (shared by 4 waves) is double-buffered in
// LDS via GLOBAL_LOAD_ASYNC_TO_LDS_B128 (ASYNCcnt) and read with ds_load_b128.
// A streams direct from L2 (96MB f16 working set is L2-resident).
// ---------------------------------------------------------------------------
__global__ __launch_bounds__(256) void gemm_spike_kernel(
    const _Float16* __restrict__ Xh, const _Float16* __restrict__ Wh,
    float* __restrict__ out) {
  constexpr int KCH   = 128;        // K halves staged per chunk
  constexpr int BROWS = 64;         // N columns (W rows) per block
  constexpr int LDB   = KCH + 8;    // padded row stride (halves): 68 words ->
                                    // lanes hit banks 4r..4r+3 = conflict-free
  __shared__ __align__(32) _Float16 Bs[2][BROWS][LDB];

  const int tid  = threadIdx.x;
  const int wave = tid >> 5;
  const int lane = tid & 31;
  const int lh   = lane & 15;      // 0..15
  const int hi   = lane >> 4;      // 0 or 1 (half-wave)

  const int rowBase = blockIdx.y * 128 + (wave & 3) * 32;   // M
  const int colBlk  = blockIdx.x * 64;                      // N block base
  const int colInW  = (wave >> 2) * 32;                     // wave's N offset in block

  const _Float16* a0p = Xh + (size_t)(rowBase + lh)      * K_DIM;
  const _Float16* a1p = Xh + (size_t)(rowBase + 16 + lh) * K_DIM;
  const int aoff = hi * 8;   // A half-wave K interleave

  // Producer: 64 rows x 128 halves = 1024 x 16B; 4 async b128 per thread.
  auto issueB = [&](int buf, int k0) {
#pragma unroll
    for (int i = 0; i < 4; ++i) {
      const int idx = i * 256 + tid;
      const int row = idx >> 4;            // 0..63
      const int c   = (idx & 15) * 8;      // halves (16B granules)
      const _Float16* src = Wh + (size_t)(colBlk + row) * K_DIM + k0 + c;
      const unsigned dst = (unsigned)(uintptr_t)&Bs[buf][row][c];
      asm volatile("global_load_async_to_lds_b128 %0, %1, off"
                   :: "v"(dst), "v"(src)
                   : "memory");
    }
  };

  v8f acc00 = {}, acc01 = {}, acc10 = {}, acc11 = {};

  issueB(0, 0);  // prefill
  for (int k0 = 0; k0 < K_DIM; k0 += KCH) {
    const int buf = (k0 / KCH) & 1;
    // my async writes into buf landed; barrier publishes all waves' writes
    asm volatile("s_wait_asynccnt 0x0" ::: "memory");
    __syncthreads();
    if (k0 + KCH < K_DIM) issueB(buf ^ 1, k0 + KCH);  // overlap next transfer

    __builtin_prefetch(a0p + k0 + KCH, 0, 0);
    __builtin_prefetch(a1p + k0 + KCH, 0, 0);

#pragma unroll
    for (int kk = 0; kk < KCH; kk += 32) {
      const int k = k0 + kk;
      v16h_u a0, a1, b0, b1;
      a0.h[0] = *(const v8h*)(a0p + k + aoff);
      a0.h[1] = *(const v8h*)(a0p + k + 16 + aoff);
      a1.h[0] = *(const v8h*)(a1p + k + aoff);
      a1.h[1] = *(const v8h*)(a1p + k + 16 + aoff);
      b0.h[0] = *(const v8h*)(&Bs[buf][colInW + lh][kk + hi * 8]);
      b0.h[1] = *(const v8h*)(&Bs[buf][colInW + lh][kk + 16 + hi * 8]);
      b1.h[0] = *(const v8h*)(&Bs[buf][colInW + 16 + lh][kk + hi * 8]);
      b1.h[1] = *(const v8h*)(&Bs[buf][colInW + 16 + lh][kk + 16 + hi * 8]);

      acc00 = __builtin_amdgcn_wmma_f32_16x16x32_f16(false, a0.v, false, b0.v,
                                                     (short)0, acc00, false, false);
      acc01 = __builtin_amdgcn_wmma_f32_16x16x32_f16(false, a0.v, false, b1.v,
                                                     (short)0, acc01, false, false);
      acc10 = __builtin_amdgcn_wmma_f32_16x16x32_f16(false, a1.v, false, b0.v,
                                                     (short)0, acc10, false, false);
      acc11 = __builtin_amdgcn_wmma_f32_16x16x32_f16(false, a1.v, false, b1.v,
                                                     (short)0, acc11, false, false);
    }
  }

  // Epilogue: fuse IF-node threshold (spike = h >= 1.0).
  // C/D layout: VGPR r -> M = r + 8*hi, N = lane%16.
  const int c0 = colBlk + colInW + lh;
  const int c1 = colBlk + colInW + 16 + lh;
  const int r0 = rowBase + hi * 8;
  const int r1 = rowBase + 16 + hi * 8;
#pragma unroll
  for (int r = 0; r < 8; ++r) {
    out[(size_t)(r0 + r) * N_DIM + c0] = (acc00[r] >= 1.0f) ? 1.0f : 0.0f;
    out[(size_t)(r0 + r) * N_DIM + c1] = (acc01[r] >= 1.0f) ? 1.0f : 0.0f;
    out[(size_t)(r1 + r) * N_DIM + c0] = (acc10[r] >= 1.0f) ? 1.0f : 0.0f;
    out[(size_t)(r1 + r) * N_DIM + c1] = (acc11[r] >= 1.0f) ? 1.0f : 0.0f;
  }
}

// ---------------------------------------------------------------------------
// Launch: ws layout = [ Xh: 8192*4096 f16 (64MB) | Wh: 4096*4096 f16 (32MB) ]
// ---------------------------------------------------------------------------
extern "C" void kernel_launch(void* const* d_in, const int* in_sizes, int n_in,
                              void* d_out, int out_size, void* d_ws, size_t ws_size,
                              hipStream_t stream) {
  (void)in_sizes; (void)n_in; (void)out_size; (void)ws_size;
  const float* x = (const float*)d_in[0];
  const float* W = (const float*)d_in[1];
  float* out = (float*)d_out;

  _Float16* Xh = (_Float16*)d_ws;
  _Float16* Wh = Xh + (size_t)M_DIM * K_DIM;

  rownorm_f16_kernel<<<M_DIM, 256, 0, stream>>>(x, Xh);
  wconv_f16_kernel<<<((size_t)N_DIM * K_DIM) / (256 * 4), 256, 0, stream>>>(W, Wh);

  dim3 grid(N_DIM / 64, M_DIM / 128);
  gemm_spike_kernel<<<grid, 256, 0, stream>>>(Xh, Wh, out);
}
